// CNNGNNModel_89515708383779
// MI455X (gfx1250) — matrix operand order repaired and checked by reading, not compiled, not run on hardware
//
#include <hip/hip_runtime.h>

// ---------------------------------------------------------------------------
// CNN+GCN model, gfx1250 (MI455X, wave32).
//
// Math: fully-connected-graph GCN with self loops => every edge norm is 1/C,
// so each GCN layer replaces node features with the per-graph mean of the
// transformed features. The GNN collapses to a per-batch MLP:
//   meanfeat[B,16] -> relu(@W1+b1)[32] -> relu(@W2+b2)[64]
//   -> concat(latent)[128] -> relu(@FC1+b1)[64] -> @FC2+b2 [2]
//
// Stage A (compute bound, ~9.7 GFLOP fp32): per-image 3x3 conv (1->8ch),
//   relu, global avg pool, 8->16 FC+relu. Pure VALU FMA (WMMA tiles would be
//   <40% utilized at K=9,N=8). Image staged into halo-padded LDS with
//   global_load_async_to_lds_b128 (no VGPR round trip, ASYNCcnt tracked);
//   conv weights forced into SGPRs (wave-uniform) via readfirstlane.
// Stage B (tiny): collapsed MLP with v_wmma_f32_16x16x32_f16, weights and
//   activations staged in LDS using the documented wave32 fragment layouts.
// ---------------------------------------------------------------------------

typedef __attribute__((ext_vector_type(16))) _Float16 v16h;
typedef __attribute__((ext_vector_type(8)))  float    v8f;

#define BATCH 256
#define CHAN  64
#define NNODE (BATCH * CHAN)
// Halo-padded image tile: interior pixel (r,c) lives at (r+1)*PITCH + (c+4).
// Row stride 72 floats => rows shift banks by 8; interior column start at +4
// keeps the 16-byte alignment required by b128 LDS writes.
#define IMG_PITCH 72

__device__ __forceinline__ float uniform_f32(float x) {
    // Force a wave-uniform value into an SGPR.
    return __int_as_float(__builtin_amdgcn_readfirstlane(__float_as_int(x)));
}

// ===========================================================================
// Stage A: conv3x3(1->8) + relu + global-avg-pool + FC(8->16) + relu
// grid = NNODE blocks, 256 threads (8 waves). Output: node_feats[NNODE][16]
// ===========================================================================
__global__ __launch_bounds__(256)
void conv_pool_fc_kernel(const float* __restrict__ x_eeg,
                         const float* __restrict__ conv_w,   // [8][9] OIHW
                         const float* __restrict__ conv_b,   // [8]
                         const float* __restrict__ cfc_w,    // [8][16]
                         const float* __restrict__ cfc_b,    // [16]
                         float* __restrict__ node_feats)     // [NNODE][16]
{
    __shared__ __align__(16) float img[66 * IMG_PITCH];  // 19008 B w/ halo
    __shared__ float red[8 * 8];   // [wave][channel]
    __shared__ float pool[8];

    const int node = blockIdx.x;
    const int tid  = threadIdx.x;
    const int lane = tid & 31;
    const int wid  = tid >> 5;

    // ---- async-stage image into LDS interior (memory -> LDS, no VGPRs) ----
    {
        const unsigned lds_base = (unsigned)(uintptr_t)(void*)&img[0];
        const float*   gsrc     = x_eeg + (size_t)node * 4096;
#pragma unroll
        for (int i = 0; i < 4; ++i) {
            int idx = i * 256 + tid;          // 16-byte chunk index
            int row = idx >> 4;               // (idx*4)/64
            int col = (idx & 15) << 2;        // (idx*4)%64
            unsigned ldsoff = lds_base +
                (unsigned)(((row + 1) * IMG_PITCH + col + 4) * 4);
            unsigned goff = (unsigned)idx * 16u;
            asm volatile("global_load_async_to_lds_b128 %0, %1, %2"
                         :: "v"(ldsoff), "v"(goff), "s"(gsrc) : "memory");
        }
    }

    // ---- zero the 1-element halo ring (regular ds stores) ----
    if (tid < IMG_PITCH) {
        img[tid] = 0.f;                              // top halo row
        img[65 * IMG_PITCH + tid] = 0.f;             // bottom halo row
    }
    if (tid < 64) {
        img[(tid + 1) * IMG_PITCH + 3]  = 0.f;       // left halo col
        img[(tid + 1) * IMG_PITCH + 68] = 0.f;       // right halo col
    }

    // ---- conv weights + bias into SGPRs (wave-uniform) ----
    float wv[72], cb[8];
#pragma unroll
    for (int i = 0; i < 72; ++i) wv[i] = uniform_f32(conv_w[i]);
#pragma unroll
    for (int i = 0; i < 8; ++i)  cb[i] = uniform_f32(conv_b[i]);

    asm volatile("s_wait_asynccnt 0x0" ::: "memory");   // drain async loads
    __syncthreads();

    // ---- 16 pixels/thread: unconditional 9-tap stencil, 8 ch, relu, pool ----
    float acc[8];
#pragma unroll
    for (int ch = 0; ch < 8; ++ch) acc[ch] = 0.f;

#pragma unroll 4
    for (int i = 0; i < 16; ++i) {
        int p   = i * 256 + tid;
        int row = p >> 6;
        int col = p & 63;
        const float* px = &img[(row + 1) * IMG_PITCH + col + 4];
        float xv[9];
#pragma unroll
        for (int dr = -1; dr <= 1; ++dr)
#pragma unroll
            for (int dc = -1; dc <= 1; ++dc)
                xv[(dr + 1) * 3 + (dc + 1)] = px[dr * IMG_PITCH + dc];
#pragma unroll
        for (int ch = 0; ch < 8; ++ch) {
            float s = cb[ch];
#pragma unroll
            for (int k = 0; k < 9; ++k) s = fmaf(wv[ch * 9 + k], xv[k], s);
            acc[ch] += fmaxf(s, 0.f);
        }
    }

    // ---- wave32 butterfly reduction, then cross-wave via LDS ----
#pragma unroll
    for (int ch = 0; ch < 8; ++ch) {
        float v = acc[ch];
#pragma unroll
        for (int off = 16; off >= 1; off >>= 1) v += __shfl_xor(v, off, 32);
        acc[ch] = v;
    }
    if (lane == 0) {
#pragma unroll
        for (int ch = 0; ch < 8; ++ch) red[wid * 8 + ch] = acc[ch];
    }
    __syncthreads();

    if (tid < 8) {
        float s = 0.f;
#pragma unroll
        for (int w = 0; w < 8; ++w) s += red[w * 8 + tid];
        pool[tid] = s * (1.0f / 4096.0f);
    }
    __syncthreads();

    // ---- FC 8->16 + relu, write node feature ----
    if (tid < 16) {
        float v = cfc_b[tid];
#pragma unroll
        for (int ch = 0; ch < 8; ++ch) v = fmaf(pool[ch], cfc_w[ch * 16 + tid], v);
        node_feats[(size_t)node * 16 + tid] = fmaxf(v, 0.f);
    }
}

// ===========================================================================
// WMMA fragment loaders (wave32, f16 16x16x32, documented VGPR layouts)
// ===========================================================================
// A (16-bit, 16xK tile): lane 0-15 = row M; VGPR v holds K pair
//   K = (v>=4 ? 16:0) + (lane>=16 ? 8:0) + (v&3)*2 + h
__device__ __forceinline__ v16h lds_load_a(const _Float16* base, int pitch,
                                           int mOff, int kOff, int lane)
{
    v16h a;
    const int M  = lane & 15;
    const int kh = (lane >> 4) * 8;
    const _Float16* row = base + (mOff + M) * pitch + kOff;
#pragma unroll
    for (int v = 0; v < 8; ++v) {
        int kb = ((v >= 4) ? 16 : 0) + (v & 3) * 2 + kh;
        a[2 * v]     = row[kb];
        a[2 * v + 1] = row[kb + 1];
    }
    return a;
}

// B (16-bit, Kx16 tile): lane 0-15 = col N; same K order as A
//   VGPR v holds K = (v>=4 ? 16:0) + (lane>=16 ? 8:0) + (v&3)*2 + h
__device__ __forceinline__ v16h lds_load_b(const _Float16* base, int pitch,
                                           int kOff, int nOff, int lane)
{
    v16h b;
    const int N  = lane & 15;
    const int kh = (lane >> 4) * 8;
#pragma unroll
    for (int v = 0; v < 8; ++v) {
        int kb = ((v >= 4) ? 16 : 0) + (v & 3) * 2 + kh;
        b[2 * v]     = base[(kOff + kb) * pitch + nOff + N];
        b[2 * v + 1] = base[(kOff + kb + 1) * pitch + nOff + N];
    }
    return b;
}

// D/C (f32 16x16): VGPR r: lanes 0-15 -> (M=r, N=lane); 16-31 -> (M=r+8)
__device__ __forceinline__ void store_d_act(_Float16* dst, int pitch,
                                            int mOff, int nOff, int lane,
                                            v8f d, const float* __restrict__ bias,
                                            bool relu)
{
    const int N  = lane & 15;
    const int mh = (lane >> 4) * 8;
    const float bv = bias[nOff + N];
#pragma unroll
    for (int r = 0; r < 8; ++r) {
        float v = d[r] + bv;
        if (relu) v = fmaxf(v, 0.f);
        dst[(mOff + mh + r) * pitch + nOff + N] = (_Float16)v;
    }
}

#define WMMA_F16(A, B, C) \
    __builtin_amdgcn_wmma_f32_16x16x32_f16(false, (A), false, (B), (short)0, (C), false, false)

// ===========================================================================
// Stage B: collapsed GCN + MLP head.
// grid = 8 blocks x 32 threads (1 wave). Each wave owns 32 batches (2 M-tiles)
// and chains 4 WMMA GEMM layers through LDS.
// ===========================================================================
__global__ __launch_bounds__(32)
void head_mlp_kernel(const float* __restrict__ node_feats,  // [NNODE][16]
                     const float* __restrict__ x_latent,    // [B][64]
                     const float* __restrict__ g1w, const float* __restrict__ g1b,
                     const float* __restrict__ g2w, const float* __restrict__ g2b,
                     const float* __restrict__ f1w, const float* __restrict__ f1b,
                     const float* __restrict__ f2w, const float* __restrict__ f2b,
                     float* __restrict__ out)                // [B][2]
{
    __shared__ _Float16 W1h[32 * 32];    // gcn1_w [16->pad32][32]
    __shared__ _Float16 W2h[32 * 64];    // gcn2_w [32][64]
    __shared__ _Float16 F1h[128 * 64];   // fc1_w  [128][64]
    __shared__ _Float16 F2h[64 * 16];    // fc2_w  [64][2->pad16]
    __shared__ _Float16 A0[32 * 32];     // meanfeat [32b][16->pad32]
    __shared__ _Float16 A1[32 * 32];     // g1 act   [32b][32]
    __shared__ _Float16 A2[32 * 128];    // concat(xg, latent) [32b][128]
    __shared__ _Float16 A3[32 * 64];     // fc1 act  [32b][64]

    const int lane  = threadIdx.x;       // 32 threads = 1 wave
    const int bBase = blockIdx.x * 32;

    // ---- stage weights into LDS as f16 (zero-pad K/N where needed) ----
    for (int i = lane; i < 32 * 32; i += 32) {                  // W1: K pad 16->32
        int k = i >> 5, n = i & 31;
        W1h[i] = (k < 16) ? (_Float16)g1w[k * 32 + n] : (_Float16)0.f;
    }
    for (int i = lane; i < 32 * 64; i += 32) W2h[i] = (_Float16)g2w[i];
    for (int i = lane; i < 128 * 64; i += 32) F1h[i] = (_Float16)f1w[i];
    for (int i = lane; i < 64 * 16; i += 32) {                  // F2: N pad 2->16
        int k = i >> 4, n = i & 15;
        F2h[i] = (n < 2) ? (_Float16)f2w[k * 2 + n] : (_Float16)0.f;
    }

    // ---- meanfeat: reduce 64 nodes x 16 feats per batch; K-pad to 32 ----
    {
        const int f     = lane & 15;
        const int chalf = lane >> 4;     // split the 64-node reduction
        for (int b = 0; b < 32; ++b) {
            const size_t nb = (size_t)(bBase + b) * CHAN;
            float s = 0.f;
#pragma unroll 8
            for (int c = chalf * 32; c < chalf * 32 + 32; ++c)
                s += node_feats[(nb + c) * 16 + f];
            s += __shfl_xor(s, 16, 32);  // combine halves; all lanes get total
            A0[b * 32 + lane] = (lane < 16) ? (_Float16)(s * (1.0f / 64.0f))
                                            : (_Float16)0.f;
        }
    }

    // ---- concat: latent half of A2 (cols 64..127) ----
    for (int i = lane; i < 32 * 64; i += 32) {
        int r = i >> 6, j = i & 63;
        A2[r * 128 + 64 + j] = (_Float16)x_latent[(size_t)(bBase + r) * 64 + j];
    }
    __syncthreads();

    // ---- L1: [32x32(K=16pad)] @ W1[32x32] -> relu -> A1[32x32] ----
#pragma unroll
    for (int mt = 0; mt < 2; ++mt) {
        v16h a = lds_load_a(A0, 32, 16 * mt, 0, lane);
#pragma unroll
        for (int nt = 0; nt < 2; ++nt) {
            v8f c = {};
            v16h b = lds_load_b(W1h, 32, 0, 16 * nt, lane);
            c = WMMA_F16(a, b, c);
            store_d_act(A1, 32, 16 * mt, 16 * nt, lane, c, g1b, true);
        }
    }
    __syncthreads();

    // ---- L2: A1[32x32] @ W2[32x64] -> relu -> A2 cols 0..63 ----
#pragma unroll
    for (int mt = 0; mt < 2; ++mt) {
        v16h a = lds_load_a(A1, 32, 16 * mt, 0, lane);
#pragma unroll
        for (int nt = 0; nt < 4; ++nt) {
            v8f c = {};
            v16h b = lds_load_b(W2h, 64, 0, 16 * nt, lane);
            c = WMMA_F16(a, b, c);
            store_d_act(A2, 128, 16 * mt, 16 * nt, lane, c, g2b, true);
        }
    }
    __syncthreads();

    // ---- L3: A2[32x128] @ FC1[128x64] -> relu -> A3[32x64] ----
#pragma unroll
    for (int mt = 0; mt < 2; ++mt) {
#pragma unroll
        for (int nt = 0; nt < 4; ++nt) {
            v8f c = {};
#pragma unroll
            for (int kt = 0; kt < 4; ++kt) {
                v16h a = lds_load_a(A2, 128, 16 * mt, 32 * kt, lane);
                v16h b = lds_load_b(F1h, 64, 32 * kt, 16 * nt, lane);
                c = WMMA_F16(a, b, c);
            }
            store_d_act(A3, 64, 16 * mt, 16 * nt, lane, c, f1b, true);
        }
    }
    __syncthreads();

    // ---- L4: A3[32x64] @ FC2[64x16(N=2 real)] + b -> out[B][2] ----
#pragma unroll
    for (int mt = 0; mt < 2; ++mt) {
        v8f c = {};
#pragma unroll
        for (int kt = 0; kt < 2; ++kt) {
            v16h a = lds_load_a(A3, 64, 16 * mt, 32 * kt, lane);
            v16h b = lds_load_b(F2h, 16, 32 * kt, 0, lane);
            c = WMMA_F16(a, b, c);
        }
        const int N  = lane & 15;
        const int mh = (lane >> 4) * 8;
        if (N < 2) {
            const float bv = f2b[N];
#pragma unroll
            for (int r = 0; r < 8; ++r)
                out[(size_t)(bBase + 16 * mt + mh + r) * 2 + N] = c[r] + bv;
        }
    }
}

// ===========================================================================
extern "C" void kernel_launch(void* const* d_in, const int* in_sizes, int n_in,
                              void* d_out, int out_size, void* d_ws, size_t ws_size,
                              hipStream_t stream) {
    const float* x_eeg  = (const float*)d_in[0];
    const float* x_lat  = (const float*)d_in[1];
    const float* conv_w = (const float*)d_in[2];
    const float* conv_b = (const float*)d_in[3];
    const float* cfc_w  = (const float*)d_in[4];
    const float* cfc_b  = (const float*)d_in[5];
    const float* g1w    = (const float*)d_in[6];
    const float* g1b    = (const float*)d_in[7];
    const float* g2w    = (const float*)d_in[8];
    const float* g2b    = (const float*)d_in[9];
    const float* f1w    = (const float*)d_in[10];
    const float* f1b    = (const float*)d_in[11];
    const float* f2w    = (const float*)d_in[12];
    const float* f2b    = (const float*)d_in[13];

    float* node_feats = (float*)d_ws;   // NNODE*16 floats = 1 MB

    conv_pool_fc_kernel<<<NNODE, 256, 0, stream>>>(
        x_eeg, conv_w, conv_b, cfc_w, cfc_b, node_feats);

    head_mlp_kernel<<<BATCH / 32, 32, 0, stream>>>(
        node_feats, x_lat, g1w, g1b, g2w, g2b, f1w, f1b, f2w, f2b,
        (float*)d_out);
}
